// TransformerBlock_16432544874977
// MI455X (gfx1250) — compile-verified, matrix operand order checked by reading, not run on hardware
//
#include <hip/hip_runtime.h>
#include <math.h>

typedef _Float16 f16;
typedef __attribute__((ext_vector_type(16))) _Float16 v16h;
typedef __attribute__((ext_vector_type(8)))  _Float16 v8h;
typedef __attribute__((ext_vector_type(8)))  float    v8f;

#define HID 256
#define SEQ 2048
#define NB  4
#define NH  4
#define HD  64
#define MLP 512
#define MTOT (NB * SEQ)   /* 8192 rows total */

#if defined(__gfx1250__) && \
    __has_builtin(__builtin_amdgcn_global_load_async_to_lds_b128) && \
    __has_builtin(__builtin_amdgcn_s_wait_asynccnt)
#define HAVE_ASYNC_LDS 1
typedef int vi4 __attribute__((vector_size(16)));
typedef __attribute__((address_space(1))) vi4 as1_vi4;
typedef __attribute__((address_space(3))) vi4 as3_vi4;
#endif

#if defined(__gfx1250__) && \
    __has_builtin(__builtin_amdgcn_tensor_load_to_lds) && \
    __has_builtin(__builtin_amdgcn_s_wait_tensorcnt)
#define HAVE_TDM 1
typedef unsigned int u32x4 __attribute__((vector_size(16)));
typedef int i32x4 __attribute__((vector_size(16)));
typedef int i32x8 __attribute__((vector_size(32)));
#endif

// ---------------------------------------------------------------------------
// helpers
// ---------------------------------------------------------------------------
__device__ __forceinline__ v8f zero8() {
  v8f z;
#pragma unroll
  for (int i = 0; i < 8; ++i) z[i] = 0.f;
  return z;
}

__device__ __forceinline__ v8f wmma32(v16h a, v16h b, v8f c) {
  // D = A(16x32 f16) * B(32x16 f16) + C(16x16 f32)
  return __builtin_amdgcn_wmma_f32_16x16x32_f16(
      /*neg_a=*/false, a, /*neg_b=*/false, b,
      /*c_mod=*/(short)0, c, /*reuse_a=*/false, /*reuse_b=*/false);
}

// A-operand fragment loader (also B-operand when given Bt = B^T, NxK row-major)
// Per ISA 7.12.2 (16-bit A 16x32):
//   lane L: row r = L&15; halfs[0..7]  = K = k0 + 8*hi + 0..7
//                         halfs[8..15] = K = k0 + 16 + 8*hi + 0..7  (hi = L>>4)
__device__ __forceinline__ v16h ld_frag(const f16* __restrict__ rowbase,
                                        int ld, int r, int hi, int k0) {
  const f16* p = rowbase + (size_t)r * ld + k0 + 8 * hi;
  union { v16h v; v8h h2[2]; } u;
  u.h2[0] = *(const v8h*)(p);
  u.h2[1] = *(const v8h*)(p + 16);
  return u.v;
}

__device__ __forceinline__ float gelu_exact(float v) {
  return 0.5f * v * (1.0f + erff(v * 0.70710678118654752f));
}

#ifdef HAVE_TDM
// Issue one TDM 2D-tile load: tile (tileK elems wide x tileN rows) of a
// row-major tensor (row length tensorK elems, stride strideK elems, 2B elems)
// starting at gaddr, into LDS byte offset lds_off (packed contiguous).
// D# layout per CDNA5 ISA.
__device__ __forceinline__ void tdm_load_2d(const void* gaddr, unsigned lds_off,
                                            unsigned tileK, unsigned tileN,
                                            unsigned tensorK, unsigned tensorN,
                                            unsigned strideK) {
  unsigned long long ga = (unsigned long long)gaddr;
  u32x4 g0;
  g0[0] = 1u;                                    // count=1 (valid user D#)
  g0[1] = lds_off;                               // lds_addr (bytes)
  g0[2] = (unsigned)(ga & 0xffffffffu);          // global_addr[31:0]
  g0[3] = (unsigned)((ga >> 32) & 0x01ffffffu)   // global_addr[56:32]
          | (2u << 30);                          // type = 2 ("image")
  i32x8 g1;
  g1[0] = (int)(1u << 16);                       // wg_mask=0, data_size=1 (2B)
  g1[1] = (int)((tensorK & 0xffffu) << 16);      // tensor_dim0[15:0] @ [63:48]
  g1[2] = (int)(((tensorK >> 16) & 0xffffu) |    // tensor_dim0[31:16]
                ((tensorN & 0xffffu) << 16));    // tensor_dim1[15:0]
  g1[3] = (int)(((tensorN >> 16) & 0xffffu) |    // tensor_dim1[31:16]
                ((tileK & 0xffffu) << 16));      // tile_dim0
  g1[4] = (int)(tileN & 0xffffu);                // tile_dim1 (tile_dim2=0)
  g1[5] = (int)strideK;                          // tensor_dim0_stride[31:0]
  g1[6] = 0;                                     // stride[47:32], dim1_stride lo
  g1[7] = 0;                                     // dim1_stride hi
  i32x4 z4 = {0, 0, 0, 0};
#if __clang_major__ >= 23
  i32x8 z8 = {0, 0, 0, 0, 0, 0, 0, 0};
  __builtin_amdgcn_tensor_load_to_lds(g0, g1, z4, z4, z8, 0);
#else
  __builtin_amdgcn_tensor_load_to_lds(g0, g1, z4, z4, 0);
#endif
}
#endif

// ---------------------------------------------------------------------------
// weight convert+transpose: src (K x N) f32 row-major -> dst (N x K) f16
// ---------------------------------------------------------------------------
__global__ void cvt_w(const float* __restrict__ src, f16* __restrict__ dst,
                      int K, int N) {
  int idx = blockIdx.x * blockDim.x + threadIdx.x;
  if (idx >= K * N) return;
  int k = idx / N, n = idx - k * N;
  dst[(size_t)n * K + k] = (f16)src[idx];
}

// ---------------------------------------------------------------------------
// LayerNorm over 256 features, one wave per token, output f16
// ---------------------------------------------------------------------------
__global__ __launch_bounds__(256) void ln_kernel(const float* __restrict__ x,
                                                 const float* __restrict__ g,
                                                 const float* __restrict__ bta,
                                                 f16* __restrict__ out,
                                                 int ntok) {
  int wid = threadIdx.x >> 5, lane = threadIdx.x & 31;
  int tok = blockIdx.x * 8 + wid;
  if (tok >= ntok) return;
  const float4* row = (const float4*)(x + (size_t)tok * HID);
  float4 a = row[lane * 2 + 0];
  float4 b = row[lane * 2 + 1];
  float vx[8] = {a.x, a.y, a.z, a.w, b.x, b.y, b.z, b.w};

  float s = 0.f;
#pragma unroll
  for (int i = 0; i < 8; ++i) s += vx[i];
#pragma unroll
  for (int m = 1; m < 32; m <<= 1) s += __shfl_xor(s, m);
  float mu = s * (1.0f / HID);

  float vs = 0.f;
#pragma unroll
  for (int i = 0; i < 8; ++i) { float d = vx[i] - mu; vs += d * d; }
#pragma unroll
  for (int m = 1; m < 32; m <<= 1) vs += __shfl_xor(vs, m);
  float rstd = rsqrtf(vs * (1.0f / HID) + 1e-6f);

  v8h o;
#pragma unroll
  for (int i = 0; i < 8; ++i) {
    int c = lane * 8 + i;
    o[i] = (f16)((vx[i] - mu) * rstd * g[c] + bta[c]);
  }
  *(v8h*)(out + (size_t)tok * HID + lane * 8) = o;
}

// ---------------------------------------------------------------------------
// WMMA GEMM:  out(M x N) = act(A(MxK,f16) @ Bt(NxK,f16)^T + bias) [+res]
//  Block = 256 threads = 8 waves; block tile 128(M) x 64(N).
//  Bt slab (64 x 32k, 4KB) double-buffered in LDS:
//   STAGE=2 -> one TDM descriptor per k-step (TENSORcnt), issued by wave 0
//   STAGE=1 -> per-thread async global->LDS b128 (ASYNCcnt)
//   fallback -> plain loads+stores
//  Each wave computes a 16x64 tile: one A-frag feeds 4 WMMAs per k-step.
//  ACT : 0 none, 1 exact GELU;  RES : add res[row*N+col] (f32)
//  OUTM: 0 -> f32 flat, 1 -> f16 flat, 2 -> f16 V-transposed (B,H,d,S)
// ---------------------------------------------------------------------------
template <int STAGE, int ACT, int RES, int OUTM>
__global__ __launch_bounds__(256) void gemm_ws(const f16* __restrict__ A,
                                               const f16* __restrict__ Bt,
                                               const float* __restrict__ bias,
                                               const float* __restrict__ res,
                                               void* __restrict__ outp,
                                               int M, int N, int K) {
  (void)M;
  __shared__ _Float16 Bs[2][64 * 32];
  int tid = threadIdx.x;
  int wid = tid >> 5, lane = tid & 31;
  int hi = lane >> 4, ln = lane & 15;
  int row0 = (blockIdx.x * 8 + wid) * 16;
  int col0 = blockIdx.y * 64;
  const f16* Ab = A + (size_t)row0 * K;

  // per-thread staging geometry: thread t moves one 16B chunk
  int srow = tid >> 2, schunk = tid & 3;
  const f16* gsrc = Bt + (size_t)(col0 + srow) * K + schunk * 8;
  int lofs = srow * 32 + schunk * 8;
  // TDM: slab base (row col0, k columns) in global, LDS byte offsets
  const f16* slab = Bt + (size_t)col0 * K;

  auto issue = [&](int buf, int k0) {
#ifdef HAVE_TDM
    if (STAGE == 2) {
      if (wid == 0) {
        unsigned ldsb = (unsigned)(size_t)(as3_vi4*)(void*)&Bs[buf][0];
        tdm_load_2d(slab + k0, ldsb, /*tileK=*/32, /*tileN=*/64,
                    /*tensorK=*/(unsigned)K, /*tensorN=*/(unsigned)N,
                    /*strideK=*/(unsigned)K);
      }
      return;
    }
#endif
#ifdef HAVE_ASYNC_LDS
    __builtin_amdgcn_global_load_async_to_lds_b128(
        (as1_vi4*)(void*)(gsrc + k0), (as3_vi4*)(void*)&Bs[buf][lofs], 0, 0);
#else
    *(v8h*)&Bs[buf][lofs] = *(const v8h*)(gsrc + k0);
#endif
  };

  auto wait_stage = [&](bool more) {
#ifdef HAVE_TDM
    if (STAGE == 2) {
      if (wid == 0) {
        if (more) __builtin_amdgcn_s_wait_tensorcnt(1);
        else      __builtin_amdgcn_s_wait_tensorcnt(0);
      }
      return;
    }
#endif
#ifdef HAVE_ASYNC_LDS
    if (more) __builtin_amdgcn_s_wait_asynccnt(1);
    else      __builtin_amdgcn_s_wait_asynccnt(0);
#endif
  };

  v8f acc[4];
#pragma unroll
  for (int t = 0; t < 4; ++t) acc[t] = zero8();

  int nk = K >> 5;
  issue(0, 0);
  for (int kb = 0; kb < nk; ++kb) {
    int k0 = kb << 5;
    int buf = kb & 1;
    if (kb + 1 < nk) {
      issue(buf ^ 1, k0 + 32);
      wait_stage(true);   // oldest (this buf) done
    } else {
      wait_stage(false);
    }
    __syncthreads();  // slab visible to all waves

    v16h a = ld_frag(Ab, K, ln, hi, k0);
#pragma unroll
    for (int t = 0; t < 4; ++t) {
      union { v16h v; v8h h2[2]; } u;
      u.h2[0] = *(const v8h*)&Bs[buf][(t * 16 + ln) * 32 + 8 * hi];
      u.h2[1] = *(const v8h*)&Bs[buf][(t * 16 + ln) * 32 + 16 + 8 * hi];
      acc[t] = wmma32(a, u.v, acc[t]);
    }
    __syncthreads();  // slab fully consumed before re-fill
  }

#pragma unroll
  for (int t = 0; t < 4; ++t) {
    int col = col0 + t * 16 + ln;
    float bv = bias[col];
#pragma unroll
    for (int i = 0; i < 8; ++i) {
      int row = row0 + i + 8 * hi;
      float v = acc[t][i] + bv;
      if (ACT) v = gelu_exact(v);
      if (RES) v += res[(size_t)row * N + col];
      if (OUTM == 0) {
        ((float*)outp)[(size_t)row * N + col] = v;
      } else if (OUTM == 1) {
        ((f16*)outp)[(size_t)row * N + col] = (f16)v;
      } else {  // V^T per (b,h): (B,H,HD,SEQ)
        int b_ = row >> 11, s = row & (SEQ - 1);
        int h_ = col >> 6, d_ = col & (HD - 1);
        ((f16*)outp)[(((size_t)(b_ * NH + h_) * HD + d_)) * SEQ + s] = (f16)v;
      }
    }
  }
}

// ---------------------------------------------------------------------------
// Flash attention: one wave per (b, h, 16-query tile).
// scores = Q K^T / 8 ; online softmax over key chunks of 32 ; ctx = P V
// P (C-layout) -> A-layout via a 16x32 LDS tile.
// ---------------------------------------------------------------------------
__global__ __launch_bounds__(32) void attn_kernel(const f16* __restrict__ Q,
                                                  const f16* __restrict__ Kf,
                                                  const f16* __restrict__ Vt,
                                                  f16* __restrict__ Att) {
  __shared__ _Float16 P[16 * 32];
  int lane = threadIdx.x;
  int hi = lane >> 4, ln = lane & 15;
  int qt = blockIdx.x, h = blockIdx.y, b = blockIdx.z;

  const f16* qptr = Q + (size_t)(b * SEQ + qt * 16) * HID + h * HD;
  v16h qa0 = ld_frag(qptr, HID, ln, hi, 0);
  v16h qa1 = ld_frag(qptr, HID, ln, hi, 32);

  float mrun[8], lrun[8];
  v8f ctx[4];
#pragma unroll
  for (int i = 0; i < 8; ++i) { mrun[i] = -INFINITY; lrun[i] = 0.f; }
#pragma unroll
  for (int t = 0; t < 4; ++t) ctx[t] = zero8();

  for (int ks = 0; ks < SEQ; ks += 32) {
    const f16* kptr0 = Kf + (size_t)(b * SEQ + ks) * HID + h * HD;
    const f16* kptr1 = kptr0 + 16 * HID;
    // prefetch next key chunk (emits global_prefetch_b8)
    __builtin_prefetch(kptr0 + 32 * HID, 0, 0);
    v8f s0 = zero8(), s1 = zero8();
    s0 = wmma32(qa0, ld_frag(kptr0, HID, ln, hi, 0), s0);
    s0 = wmma32(qa1, ld_frag(kptr0, HID, ln, hi, 32), s0);
    s1 = wmma32(qa0, ld_frag(kptr1, HID, ln, hi, 0), s1);
    s1 = wmma32(qa1, ld_frag(kptr1, HID, ln, hi, 32), s1);

    float scale[8];
#pragma unroll
    for (int i = 0; i < 8; ++i) {
      float a0 = s0[i] * 0.125f;   // 1/sqrt(64)
      float a1 = s1[i] * 0.125f;
      float t = fmaxf(a0, a1);
#pragma unroll
      for (int m = 1; m < 16; m <<= 1) t = fmaxf(t, __shfl_xor(t, m));
      float mnew = fmaxf(mrun[i], t);
      float p0 = expf(a0 - mnew);
      float p1 = expf(a1 - mnew);
      float rs = p0 + p1;
#pragma unroll
      for (int m = 1; m < 16; m <<= 1) rs += __shfl_xor(rs, m);
      scale[i] = expf(mrun[i] - mnew);
      lrun[i] = lrun[i] * scale[i] + rs;
      mrun[i] = mnew;
      s0[i] = p0;
      s1[i] = p1;
    }
#pragma unroll
    for (int t = 0; t < 4; ++t)
#pragma unroll
      for (int i = 0; i < 8; ++i) ctx[t][i] *= scale[i];

    // C-layout -> LDS (16 rows x 32 keys, row-major f16)
#pragma unroll
    for (int i = 0; i < 8; ++i) {
      P[(i + 8 * hi) * 32 + ln]      = (f16)s0[i];
      P[(i + 8 * hi) * 32 + 16 + ln] = (f16)s1[i];
    }
    asm volatile("s_wait_dscnt 0" ::: "memory");

    // A-layout read of P from LDS
    union { v16h v; v8h h2[2]; } up;
    up.h2[0] = *(const v8h*)(&P[ln * 32 + 8 * hi]);
    up.h2[1] = *(const v8h*)(&P[ln * 32 + 16 + 8 * hi]);
    v16h pa = up.v;

#pragma unroll
    for (int t = 0; t < 4; ++t) {
      const f16* vptr = Vt + ((size_t)(b * NH + h) * HD + t * 16) * SEQ + ks;
      __builtin_prefetch(vptr + 32, 0, 0);
      ctx[t] = wmma32(pa, ld_frag(vptr, SEQ, ln, hi, 0), ctx[t]);
    }
  }

#pragma unroll
  for (int t = 0; t < 4; ++t)
#pragma unroll
    for (int i = 0; i < 8; ++i) {
      float val = ctx[t][i] / lrun[i];
      Att[(size_t)(b * SEQ + qt * 16 + i + 8 * hi) * HID + h * HD + t * 16 + ln] =
          (f16)val;
    }
}

// ---------------------------------------------------------------------------
// host launcher
// ---------------------------------------------------------------------------
extern "C" void kernel_launch(void* const* d_in, const int* in_sizes, int n_in,
                              void* d_out, int out_size, void* d_ws,
                              size_t ws_size, hipStream_t stream) {
  (void)in_sizes; (void)n_in; (void)out_size; (void)ws_size;
  const float* x     = (const float*)d_in[0];
  const float* ln1_g = (const float*)d_in[1];
  const float* ln1_b = (const float*)d_in[2];
  const float* Wq    = (const float*)d_in[3];
  const float* bq    = (const float*)d_in[4];
  const float* Wk    = (const float*)d_in[5];
  const float* bk    = (const float*)d_in[6];
  const float* Wv    = (const float*)d_in[7];
  const float* bv    = (const float*)d_in[8];
  const float* Wo    = (const float*)d_in[9];
  const float* bo    = (const float*)d_in[10];
  const float* ln2_g = (const float*)d_in[11];
  const float* ln2_b = (const float*)d_in[12];
  const float* W1    = (const float*)d_in[13];
  const float* b1    = (const float*)d_in[14];
  const float* W2    = (const float*)d_in[15];
  const float* b2    = (const float*)d_in[16];

  char* ws = (char*)d_ws;
  size_t off = 0;
  auto take = [&](size_t bytes) -> void* {
    void* p = ws + off;
    off += (bytes + 255) & ~(size_t)255;
    return p;
  };

  f16*   hf  = (f16*)take((size_t)MTOT * HID * 2);
  f16*   Wqt = (f16*)take((size_t)HID * HID * 2);
  f16*   Wkt = (f16*)take((size_t)HID * HID * 2);
  f16*   Wvt = (f16*)take((size_t)HID * HID * 2);
  f16*   Wot = (f16*)take((size_t)HID * HID * 2);
  f16*   W1t = (f16*)take((size_t)MLP * HID * 2);
  f16*   W2t = (f16*)take((size_t)HID * MLP * 2);
  f16*   Qf  = (f16*)take((size_t)MTOT * HID * 2);
  f16*   Kfb = (f16*)take((size_t)MTOT * HID * 2);
  f16*   Vtb = (f16*)take((size_t)MTOT * HID * 2);
  f16*   Atb = (f16*)take((size_t)MTOT * HID * 2);
  float* x1  = (float*)take((size_t)MTOT * HID * 4);
  f16*   h2  = (f16*)take((size_t)MTOT * HID * 2);
  f16*   m1  = (f16*)take((size_t)MTOT * MLP * 2);

  // 1) weight transpose/convert to f16
  cvt_w<<<(HID * HID + 255) / 256, 256, 0, stream>>>(Wq, Wqt, HID, HID);
  cvt_w<<<(HID * HID + 255) / 256, 256, 0, stream>>>(Wk, Wkt, HID, HID);
  cvt_w<<<(HID * HID + 255) / 256, 256, 0, stream>>>(Wv, Wvt, HID, HID);
  cvt_w<<<(HID * HID + 255) / 256, 256, 0, stream>>>(Wo, Wot, HID, HID);
  cvt_w<<<(HID * MLP + 255) / 256, 256, 0, stream>>>(W1, W1t, HID, MLP);
  cvt_w<<<(MLP * HID + 255) / 256, 256, 0, stream>>>(W2, W2t, MLP, HID);

  // 2) LN1
  ln_kernel<<<MTOT / 8, 256, 0, stream>>>(x, ln1_g, ln1_b, hf, MTOT);

  // 3) Q/K/V projections (async-LDS staging; V stored transposed per head)
  dim3 gqkv(MTOT / 128, HID / 64);
  gemm_ws<1, 0, 0, 1><<<gqkv, 256, 0, stream>>>(hf, Wqt, bq, nullptr, Qf, MTOT, HID, HID);
  gemm_ws<1, 0, 0, 1><<<gqkv, 256, 0, stream>>>(hf, Wkt, bk, nullptr, Kfb, MTOT, HID, HID);
  gemm_ws<1, 0, 0, 2><<<gqkv, 256, 0, stream>>>(hf, Wvt, bv, nullptr, Vtb, MTOT, HID, HID);

  // 4) attention
  attn_kernel<<<dim3(SEQ / 16, NH, NB), 32, 0, stream>>>(Qf, Kfb, Vtb, Atb);

  // 5) output projection + residual -> x1 (f32)   (TDM staging)
  gemm_ws<2, 0, 1, 0><<<gqkv, 256, 0, stream>>>(Atb, Wot, bo, x, x1, MTOT, HID, HID);

  // 6) LN2
  ln_kernel<<<MTOT / 8, 256, 0, stream>>>(x1, ln2_g, ln2_b, h2, MTOT);

  // 7) MLP1: h2 @ W1 + b1, GELU, f16   (TDM staging)
  dim3 gm1(MTOT / 128, MLP / 64);
  gemm_ws<2, 1, 0, 1><<<gm1, 256, 0, stream>>>(h2, W1t, b1, nullptr, m1, MTOT, MLP, HID);

  // 8) MLP2: gelu(m1 @ W2 + b2) + x1 -> d_out (f32)   (TDM staging)
  dim3 gm2(MTOT / 128, HID / 64);
  gemm_ws<2, 1, 1, 0><<<gm2, 256, 0, stream>>>(m1, W2t, b2, x1, d_out, MTOT, HID, MLP);
}
